// CombineGraph_75299366633942
// MI455X (gfx1250) — matrix operand order repaired and checked by reading, not compiled, not run on hardware
//
#include <hip/hip_runtime.h>
#include <hip/hip_bf16.h>
#include <math.h>

#define B_  128
#define S_  50
#define L_  50
#define D_  128
#define H_  128
#define E_  128
#define SN_ 10
#define G3  (3 * H_)    // 384
#define WSTRIDE 136     // padded LDS row stride (halves): bank-conflict-free, 16B-aligned frags

typedef __attribute__((ext_vector_type(16))) _Float16 v16h;
typedef __attribute__((ext_vector_type(8)))  _Float16 v8h;
typedef __attribute__((ext_vector_type(8)))  float    v8f;

__device__ __forceinline__ float sigmoidf_(float x) { return 1.0f / (1.0f + expf(-x)); }

// ---------------- WMMA fragment loaders (wave32, 16x16x32 f16) ----------------
// A-matrix 16x32 (MxK), f32 source converted to f16 (contiguous k order, same as the B loaders):
// lane: m = lane&15, hi = lane>>4; element e:
// K = 16*hi + e
__device__ __forceinline__ v16h load_a_frag(const float* __restrict__ A, int lda,
                                            int base_row, int kbase, int lane) {
  int m  = base_row + (lane & 15);
  int hi = lane >> 4;
  v16h a;
#pragma unroll
  for (int e = 0; e < 16; ++e) {
    int k = 16 * hi + e;
    a[e] = (_Float16)A[(size_t)m * lda + kbase + k];
  }
  return a;
}

// B-matrix 32x16 (KxN) where f32 source is stored transposed: B[k,n] = Bt[n*ldb + k].
// element e: K = kbase + e + 16*hi  ->  16 K-contiguous floats per lane (vectorizes).
__device__ __forceinline__ v16h load_b_frag_tr(const float* __restrict__ Bt, int ldb,
                                               int kbase, int base_col, int lane) {
  int n  = base_col + (lane & 15);
  int hi = lane >> 4;
  const float* p = Bt + (size_t)n * ldb + kbase + 16 * hi;
  v16h b;
#pragma unroll
  for (int e = 0; e < 16; ++e) b[e] = (_Float16)p[e];
  return b;
}

// B fragment from an LDS-staged f16 weight tile stored [n_local][k] with row stride WSTRIDE.
// Fragment elements are K-contiguous -> two aligned 16B ds_load_b128 per lane.
__device__ __forceinline__ v16h load_b_frag_lds(const _Float16* Wl,
                                                int kbase, int n_local_base, int lane) {
  int nl = n_local_base + (lane & 15);
  int hi = lane >> 4;
  const _Float16* p = Wl + nl * WSTRIDE + kbase + 16 * hi;
  v8h lo = *(const v8h*)(p);
  v8h hh = *(const v8h*)(p + 8);
  return __builtin_shufflevector(lo, hh, 0, 1, 2, 3, 4, 5, 6, 7,
                                 8, 9, 10, 11, 12, 13, 14, 15);
}

// ---------------- K1: fused 2-hop global aggregation + embedding gather ----------------
// One block per (b,s); 128 threads own the 128 D-channels. ~327MB of L2-resident gathers;
// prefetch the next j-group's index/weight rows to overlap the dependent index chase.
__global__ void global_agg_kernel(const int* __restrict__ inputs,
                                  const int* __restrict__ adj_all,
                                  const float* __restrict__ num,
                                  const float* __restrict__ embedding,
                                  float* __restrict__ h_gather,
                                  float* __restrict__ h_global) {
  int bs = blockIdx.x;          // b*S + s
  int d  = threadIdx.x;         // 0..127
  int root = inputs[bs];

  h_gather[(size_t)bs * D_ + d] = embedding[(size_t)root * D_ + d];

  // softmax over hop-0 weights num[root, 0..9]
  float w0[SN_];
  float m0 = -1e30f;
#pragma unroll
  for (int j = 0; j < SN_; ++j) { w0[j] = num[root * SN_ + j]; m0 = fmaxf(m0, w0[j]); }
  float s0 = 0.f;
#pragma unroll
  for (int j = 0; j < SN_; ++j) { w0[j] = expf(w0[j] - m0); s0 += w0[j]; }
  float inv0 = 1.f / s0;

  float acc = 0.f;
  for (int j = 0; j < SN_; ++j) {
    int n1 = adj_all[root * SN_ + j];
    if (j + 1 < SN_) {                       // prefetch next dependent gather group
      int n1n = adj_all[root * SN_ + j + 1];
      __builtin_prefetch((const void*)(num + (size_t)n1n * SN_), 0, 0);
      __builtin_prefetch((const void*)(adj_all + (size_t)n1n * SN_), 0, 0);
    }
    float w1[SN_];
    float m1 = -1e30f;
#pragma unroll
    for (int k = 0; k < SN_; ++k) { w1[k] = num[n1 * SN_ + k]; m1 = fmaxf(m1, w1[k]); }
    float s1 = 0.f;
#pragma unroll
    for (int k = 0; k < SN_; ++k) { w1[k] = expf(w1[k] - m1); s1 += w1[k]; }
    float inv1 = 1.f / s1;
    float inner = 0.f;
#pragma unroll
    for (int k = 0; k < SN_; ++k) {
      int n2 = adj_all[n1 * SN_ + k];
      inner += w1[k] * embedding[(size_t)n2 * D_ + d];
    }
    inner *= inv1;                               // hop-2 aggregate (agg1)
    acc += (w0[j] * inv0) * fmaxf(inner, 0.f);   // relu then hop-1 weight
  }
  h_global[(size_t)bs * D_ + d] = fmaxf(acc, 0.f);
}

// ---------------- K2: local branch — symmetric degree norm + 50x50 @ 50x128 ----------------
__global__ void local_kernel(const float* __restrict__ adj,
                             const float* __restrict__ h_gather,
                             float* __restrict__ h_local) {
  __shared__ float adjL[S_ * S_];
  __shared__ float rowsum[S_];
  __shared__ float colsum[S_];
  __shared__ float hs[S_ * D_];
  int b = blockIdx.x, tid = threadIdx.x;

  for (int idx = tid; idx < S_ * S_; idx += blockDim.x)
    adjL[idx] = adj[(size_t)b * S_ * S_ + idx];
  __syncthreads();

  if (tid < S_) {                 // row[b,x] = sum_j adj[b,x,j]
    float s = 0.f;
    for (int j = 0; j < S_; ++j) s += adjL[tid * S_ + j];
    rowsum[tid] = s;
  } else if (tid >= 64 && tid < 64 + S_) {   // col[b,x] = sum_i adj[b,i,x]
    int i = tid - 64;
    float s = 0.f;
    for (int k = 0; k < S_; ++k) s += adjL[k * S_ + i];
    colsum[i] = s;
  }
  __syncthreads();

  // W[i,j] = adj[i,j] * rsqrt(col[i]*row[j]) where positive
  for (int idx = tid; idx < S_ * S_; idx += blockDim.x) {
    int i = idx / S_, j = idx % S_;
    float p = colsum[i] * rowsum[j];
    adjL[idx] = (p > 0.f) ? adjL[idx] * rsqrtf(p) : 0.f;
  }
  for (int idx = tid; idx < S_ * D_; idx += blockDim.x)
    hs[idx] = h_gather[(size_t)b * S_ * D_ + idx];
  __syncthreads();

  for (int idx = tid; idx < S_ * D_; idx += blockDim.x) {
    int i = idx >> 7, d = idx & 127;
    float acc = 0.f;
#pragma unroll 10
    for (int j = 0; j < S_; ++j) acc += adjL[i * S_ + j] * hs[j * D_ + d];
    h_local[(size_t)b * S_ * D_ + idx] = acc;
  }
}

// ---------------- K3: gated fusion (two WMMA GEMMs [6400,128]@[128,128] + gate) ----------------
// grid (400, 2), 128 threads (4 waves). Each block stages a 64-column slice of ws and wg
// into LDS as f16 [n][k]; waves each own one 16-col n-tile and compute both GEMMs.
__global__ void fusion_kernel(const float* __restrict__ h_local,
                              const float* __restrict__ h_global,
                              const float* __restrict__ ws,
                              const float* __restrict__ wg,
                              float* __restrict__ out) {
  __shared__ __attribute__((aligned(16))) _Float16 wsL[64 * WSTRIDE];
  __shared__ __attribute__((aligned(16))) _Float16 wgL[64 * WSTRIDE];
  int tid = threadIdx.x;
  int wave = tid >> 5, lane = tid & 31;
  int base_row = blockIdx.x * 16;          // 400 m-tiles
  int col0 = blockIdx.y * 64;              // column half

  // stage (coalesced over n): ws/wg are [k][n] row-major -> store transposed [n][k]
  for (int idx = tid; idx < 64 * 128; idx += 128) {
    int nl = idx & 63, k = idx >> 6;
    wsL[nl * WSTRIDE + k] = (_Float16)ws[(size_t)k * D_ + col0 + nl];
    wgL[nl * WSTRIDE + k] = (_Float16)wg[(size_t)k * D_ + col0 + nl];
  }
  __syncthreads();

  int nlb = wave * 16;                     // this wave's n-tile within the half
  v8f acc1 = {}, acc2 = {};
#pragma unroll
  for (int ks = 0; ks < 4; ++ks) {
    int kbase = ks * 32;
    v16h a1 = load_a_frag(h_local, D_, base_row, kbase, lane);
    v16h b1 = load_b_frag_lds(wsL, kbase, nlb, lane);
    acc1 = __builtin_amdgcn_wmma_f32_16x16x32_f16(false, a1, false, b1, (short)0, acc1, false, false);
    v16h a2 = load_a_frag(h_global, D_, base_row, kbase, lane);
    v16h b2 = load_b_frag_lds(wgL, kbase, nlb, lane);
    acc2 = __builtin_amdgcn_wmma_f32_16x16x32_f16(false, a2, false, b2, (short)0, acc2, false, false);
  }
  int hi = lane >> 4;
  int ncol = col0 + nlb + (lane & 15);
#pragma unroll
  for (int i = 0; i < 8; ++i) {
    int row = base_row + i + 8 * hi;
    float hl = h_local[(size_t)row * D_ + ncol];
    float hg = h_global[(size_t)row * D_ + ncol];
    float rg = sigmoidf_(acc1[i] + acc2[i]);
    out[(size_t)row * D_ + ncol] = hg * rg + hl - hl * rg;
  }
}

// ---------------- K4: seq embedding gather + h_state zero ----------------
__global__ void embs_gather_kernel(const int* __restrict__ seq,
                                   const float* __restrict__ emb_w,
                                   float* __restrict__ embs) {
  int idx = blockIdx.x * blockDim.x + threadIdx.x;   // over L*B*E
  if (idx >= L_ * B_ * E_) return;
  int e = idx & 127, tb = idx >> 7;
  embs[idx] = emb_w[(size_t)seq[tb] * E_ + e];
}

__global__ void zero_kernel(float* __restrict__ p, int n) {
  int i = blockIdx.x * blockDim.x + threadIdx.x;
  if (i < n) p[i] = 0.f;
}

// ---------------- K5: GRU input gates gi = embs @ w_ih^T + b_ih  ([6400,128]@[128,384]) ----------------
// grid (400, 3), 256 threads (8 waves). w_ih is [n][k] row-major: straight coalesced f16
// stage of 128 rows into LDS; 8 waves share the staged tile (one 16-col n-tile each).
__global__ void gi_kernel(const float* __restrict__ embs,
                          const float* __restrict__ w_ih,
                          const float* __restrict__ b_ih,
                          float* __restrict__ gi_all) {
  __shared__ __attribute__((aligned(16))) _Float16 wL[128 * WSTRIDE];
  int tid = threadIdx.x;
  int wave = tid >> 5, lane = tid & 31;
  int base_row = blockIdx.x * 16;                 // 400 m-tiles
  int n0 = blockIdx.y * 128;                      // 128-column slice of the 384 gates

  for (int idx = tid; idx < 128 * 128; idx += 256) {
    int nl = idx >> 7, k = idx & 127;
    wL[nl * WSTRIDE + k] = (_Float16)w_ih[(size_t)(n0 + nl) * E_ + k];
  }
  __syncthreads();

  int nlb = wave * 16;
  v8f acc = {};
#pragma unroll
  for (int ks = 0; ks < 4; ++ks) {
    v16h a = load_a_frag(embs, E_, base_row, ks * 32, lane);
    v16h b = load_b_frag_lds(wL, ks * 32, nlb, lane);
    acc = __builtin_amdgcn_wmma_f32_16x16x32_f16(false, a, false, b, (short)0, acc, false, false);
  }
  int hi = lane >> 4;
  int ncol = n0 + nlb + (lane & 15);
  float bias = b_ih[ncol];
#pragma unroll
  for (int i = 0; i < 8; ++i) {
    int row = base_row + i + 8 * hi;
    gi_all[(size_t)row * G3 + ncol] = acc[i] + bias;
  }
}

// ---------------- K6: one GRU time step (WMMA recurrence gh = h @ w_hh^T, fused gates) ----------------
// A fragments (this block's 16 hidden-state rows) hoisted to registers and reused for all
// 3 n-tiles per wave; w_hh fragments are K-contiguous global loads (vectorize to b128).
__global__ void gru_step_kernel(const float* __restrict__ gi_all,
                                const float* __restrict__ w_hh,
                                const float* __restrict__ b_hh,
                                const int* __restrict__ lengths,
                                float* __restrict__ h_state,
                                float* __restrict__ gru_out,
                                int t) {
  __shared__ float gh[16 * G3];
  int wave = threadIdx.x >> 5, lane = threadIdx.x & 31;
  int mtile = blockIdx.x;                          // 8 tiles over B=128
  int hi = lane >> 4;

  v16h afrag[4];
#pragma unroll
  for (int ks = 0; ks < 4; ++ks)
    afrag[ks] = load_a_frag(h_state, H_, mtile * 16, ks * 32, lane);

  for (int nt = wave * 3; nt < wave * 3 + 3; ++nt) {   // 24 n-tiles / 8 waves
    v8f acc = {};
#pragma unroll
    for (int ks = 0; ks < 4; ++ks) {
      v16h b = load_b_frag_tr(w_hh, H_, ks * 32, nt * 16, lane);
      acc = __builtin_amdgcn_wmma_f32_16x16x32_f16(false, afrag[ks], false, b, (short)0, acc, false, false);
    }
    int ncol = nt * 16 + (lane & 15);
    float bias = b_hh[ncol];
#pragma unroll
    for (int i = 0; i < 8; ++i) {
      int m = i + 8 * hi;
      gh[m * G3 + ncol] = acc[i] + bias;
    }
  }
  __syncthreads();

  for (int idx = threadIdx.x; idx < 16 * H_; idx += blockDim.x) {
    int m = idx >> 7, hc = idx & 127;
    int row = mtile * 16 + m;                      // batch index
    const float* gi = gi_all + ((size_t)t * B_ + row) * G3;
    float i_r = gi[hc], i_z = gi[H_ + hc], i_n = gi[2 * H_ + hc];
    float g_r = gh[m * G3 + hc], g_z = gh[m * G3 + H_ + hc], g_n = gh[m * G3 + 2 * H_ + hc];
    float hold = h_state[(size_t)row * H_ + hc];
    float r = sigmoidf_(i_r + g_r);
    float z = sigmoidf_(i_z + g_z);
    float n = tanhf(i_n + r * g_n);
    float hnew = (1.f - z) * n + z * hold;
    bool valid = t < lengths[row];
    h_state[(size_t)row * H_ + hc] = valid ? hnew : hold;
    gru_out[((size_t)row * L_ + t) * H_ + hc] = valid ? hnew : 0.f;
  }
}

// ---------------- K7: q2 = ht @ a2_w^T (tiny) ----------------
__global__ void q2_kernel(const float* __restrict__ h_state,
                          const float* __restrict__ a2_w,
                          float* __restrict__ q2) {
  int b = blockIdx.x, h = threadIdx.x;
  float s = 0.f;
#pragma unroll 16
  for (int k = 0; k < H_; ++k) s += h_state[(size_t)b * H_ + k] * a2_w[(size_t)h * H_ + k];
  q2[(size_t)b * H_ + h] = s;
}

// ---------------- K8: q1act = sigmoid(gru_out @ a1_w^T + mask*q2)  (WMMA, LDS-staged a1_w) ----------------
__global__ void q1act_kernel(const float* __restrict__ gru_out,
                             const float* __restrict__ a1_w,
                             const float* __restrict__ q2,
                             const int* __restrict__ seq,
                             float* __restrict__ q1a) {
  __shared__ __attribute__((aligned(16))) _Float16 wL[128 * WSTRIDE];
  int tid = threadIdx.x;
  int wave = tid >> 5, lane = tid & 31;
  int base_row = blockIdx.x * 16;   // 400 m-tiles over B*L

  for (int idx = tid; idx < 128 * 128; idx += 256) {
    int nl = idx >> 7, k = idx & 127;
    wL[nl * WSTRIDE + k] = (_Float16)a1_w[(size_t)nl * H_ + k];
  }
  __syncthreads();

  int nlb = wave * 16;              // 8 n-tiles over H
  v8f acc = {};
#pragma unroll
  for (int ks = 0; ks < 4; ++ks) {
    v16h a = load_a_frag(gru_out, H_, base_row, ks * 32, lane);
    v16h b = load_b_frag_lds(wL, ks * 32, nlb, lane);
    acc = __builtin_amdgcn_wmma_f32_16x16x32_f16(false, a, false, b, (short)0, acc, false, false);
  }
  int hi = lane >> 4;
  int ncol = nlb + (lane & 15);
#pragma unroll
  for (int i = 0; i < 8; ++i) {
    int row = base_row + i + 8 * hi;
    int bb = row / L_, l = row % L_;
    float mk = (seq[l * B_ + bb] > 0) ? 1.f : 0.f;
    q1a[(size_t)row * H_ + ncol] = sigmoidf_(acc[i] + mk * q2[(size_t)bb * H_ + ncol]);
  }
}

// ---------------- K9: alpha, c_local, inverse-permutation scatter ----------------
__global__ void readout_kernel(const float* __restrict__ q1a,
                               const float* __restrict__ gru_out,
                               const float* __restrict__ vt,
                               const int* __restrict__ order,
                               float* __restrict__ bt_out) {
  __shared__ float alphaL[L_];
  int b = blockIdx.x, tid = threadIdx.x;
  if (tid < L_) {
    float s = 0.f;
    const float* row = q1a + ((size_t)b * L_ + tid) * H_;
#pragma unroll 16
    for (int k = 0; k < H_; ++k) s += row[k] * vt[k];
    alphaL[tid] = s;
  }
  __syncthreads();
  float c = 0.f;
  for (int l = 0; l < L_; ++l)
    c += alphaL[l] * gru_out[((size_t)b * L_ + l) * H_ + tid];
  // b_t[order[b]] = c_local[b]  (argsort of a permutation == inverse permutation)
  bt_out[(size_t)order[b] * H_ + tid] = c;
}

extern "C" void kernel_launch(void* const* d_in, const int* in_sizes, int n_in,
                              void* d_out, int out_size, void* d_ws, size_t ws_size,
                              hipStream_t stream) {
  (void)in_sizes; (void)n_in; (void)out_size; (void)ws_size;
  const int*   inputs    = (const int*)d_in[0];
  const float* adj       = (const float*)d_in[1];
  // d_in[2] mask_item, d_in[3] item: unused by the reference forward
  const int*   seq       = (const int*)d_in[4];
  const int*   lengths   = (const int*)d_in[5];
  const int*   order     = (const int*)d_in[6];
  const int*   adj_all   = (const int*)d_in[7];
  const float* num       = (const float*)d_in[8];
  const float* embedding = (const float*)d_in[9];
  const float* emb_w     = (const float*)d_in[10];
  const float* ws        = (const float*)d_in[11];
  const float* wg        = (const float*)d_in[12];
  const float* w_ih      = (const float*)d_in[13];
  const float* w_hh      = (const float*)d_in[14];
  const float* b_ih      = (const float*)d_in[15];
  const float* b_hh      = (const float*)d_in[16];
  const float* a1_w      = (const float*)d_in[17];
  const float* a2_w      = (const float*)d_in[18];
  const float* vt_w      = (const float*)d_in[19];
  float* out = (float*)d_out;

  float* wsf    = (float*)d_ws;
  float* h_g    = wsf;                 //  819200  embedding[inputs]
  float* h_glb  = wsf + 819200;        //  819200  h_global
  float* h_loc  = wsf + 1638400;       //  819200  h_local
  float* embs   = wsf + 2457600;       //  819200  emb_w[seq]
  float* gi     = wsf + 3276800;       // 2457600  input gates [L*B,384]
  float* gru_o  = wsf + 5734400;       //  819200  gru_out [B,L,H]
  float* h_st   = wsf + 6553600;       //   16384  hidden state
  float* q2     = wsf + 6569984;       //   16384
  float* q1a    = wsf + 6586368;       //  819200

  // graph branch -> d_out[0 .. B*S*D)
  global_agg_kernel<<<B_ * S_, D_, 0, stream>>>(inputs, adj_all, num, embedding, h_g, h_glb);
  local_kernel<<<B_, 256, 0, stream>>>(adj, h_g, h_loc);
  fusion_kernel<<<dim3(400, 2), 128, 0, stream>>>(h_loc, h_glb, ws, wg, out);

  // GRU branch -> d_out[B*S*D ..)
  embs_gather_kernel<<<(L_ * B_ * E_) / 256, 256, 0, stream>>>(seq, emb_w, embs);
  zero_kernel<<<(B_ * H_ + 255) / 256, 256, 0, stream>>>(h_st, B_ * H_);
  gi_kernel<<<dim3(400, 3), 256, 0, stream>>>(embs, w_ih, b_ih, gi);
  for (int t = 0; t < L_; ++t)
    gru_step_kernel<<<8, 256, 0, stream>>>(gi, w_hh, b_hh, lengths, h_st, gru_o, t);
  q2_kernel<<<B_, H_, 0, stream>>>(h_st, a2_w, q2);
  q1act_kernel<<<400, 256, 0, stream>>>(gru_o, a1_w, q2, seq, q1a);
  readout_kernel<<<B_, H_, 0, stream>>>(q1a, gru_o, vt_w, order, out + B_ * S_ * D_);
}
